// RBNPlus_48627619725445
// MI455X (gfx1250) — compile-verified, matrix operand order checked by reading, not run on hardware
//
#include <hip/hip_runtime.h>
#include <hip/hip_bf16.h>
#include <math.h>

#define B_ 2
#define N_ 2048
#define C_ 256
#define K_ 64
#define H_ 4
#define DH_ 32
#define DM_ 128
#define DI_ 16
#define L_ 3
#define S_ 64
#define OUT_ 10

typedef __attribute__((ext_vector_type(16))) _Float16     v16h;
typedef __attribute__((ext_vector_type(8)))  float        v8f;
typedef __attribute__((ext_vector_type(4)))  unsigned int v4u;
typedef __attribute__((ext_vector_type(8)))  int          v8i;
typedef __attribute__((ext_vector_type(4)))  int          v4i;

__device__ __forceinline__ float gelu_exact(float x) {
    return 0.5f * x * (1.0f + erff(x * 0.7071067811865475f));
}

// -------------------------------------------------------------------------
// Fragment loader: 16x32 f16 A/B fragment from a row-major f32 tile, using
// 128-bit loads (all addresses 16B aligned: ld multiple of 32, kb in {0,8}).
// ISA layout (16-bit A 16x32): lanes 0-15 -> M=lane, K=0..7 and 16..23;
// lanes 16-31 -> M=lane-16, K=8..15 and 24..31.
// -------------------------------------------------------------------------
__device__ __forceinline__ v16h load_frag_f32(const float* __restrict__ base,
                                              int ld, int lane) {
    const int row = lane & 15;
    const int kb  = (lane < 16) ? 0 : 8;
    const float* p = base + row * ld + kb;
    const float4 x0 = *(const float4*)(p);
    const float4 x1 = *(const float4*)(p + 4);
    const float4 y0 = *(const float4*)(p + 16);
    const float4 y1 = *(const float4*)(p + 20);
    v16h f;
    f[0]  = (_Float16)x0.x; f[1]  = (_Float16)x0.y;
    f[2]  = (_Float16)x0.z; f[3]  = (_Float16)x0.w;
    f[4]  = (_Float16)x1.x; f[5]  = (_Float16)x1.y;
    f[6]  = (_Float16)x1.z; f[7]  = (_Float16)x1.w;
    f[8]  = (_Float16)y0.x; f[9]  = (_Float16)y0.y;
    f[10] = (_Float16)y0.z; f[11] = (_Float16)y0.w;
    f[12] = (_Float16)y1.x; f[13] = (_Float16)y1.y;
    f[14] = (_Float16)y1.z; f[15] = (_Float16)y1.w;
    return f;
}

// -------------------------------------------------------------------------
// Generic WMMA linear: Out[M,Nc] = act(A[M,Kd] * W[Nc,Kd]^T + bias)
// One wave per 16x16 output tile. The 16-row weight tile is staged into LDS
// by the Tensor Data Mover (TDM) as a 2-D D# tile, then consumed via
// ds_load-fed B fragments. act: 0=none, 1=gelu.
// -------------------------------------------------------------------------
__global__ void wmma_linear_kernel(const float* __restrict__ A,
                                   const float* __restrict__ W,
                                   const float* __restrict__ bias,
                                   float* __restrict__ Out,
                                   int M, int Kd, int Nc, int act) {
    const int lane = threadIdx.x;
    const int nt = blockIdx.x * 16;
    const int mt = blockIdx.y * 16;

    __shared__ float sW[16 * 512];  // max Kd = 512 -> 32 KB

    // ---- TDM: load 16 x Kd f32 weight tile (rows nt..nt+15) into LDS ----
    {
        const unsigned long long ga =
            (unsigned long long)(uintptr_t)(W + (size_t)nt * Kd);
        const unsigned lds_off = (unsigned)(uintptr_t)(&sW[0]);
        const unsigned dim0 = (unsigned)Kd;   // tensor width (elements)
        const unsigned dim1 = 16u;            // tensor height
        const unsigned long long stride0 = (unsigned long long)Kd;
        v4u g0;
        g0[0] = 1u;                                    // count=1 (valid D#)
        g0[1] = lds_off;                               // lds_addr
        g0[2] = (unsigned)(ga & 0xFFFFFFFFu);          // global_addr[31:0]
        g0[3] = (unsigned)((ga >> 32) & 0x01FFFFFFu)   // global_addr[56:32]
                | 0x80000000u;                         // type=2 ("image")
        v8i g1;
        g1[0] = (int)(2u << 16);                       // data_size=4B
        g1[1] = (int)((dim0 & 0xFFFFu) << 16);         // tensor_dim0[15:0]
        g1[2] = (int)((dim0 >> 16) | ((dim1 & 0xFFFFu) << 16));  // dim0 hi | dim1 lo
        g1[3] = (int)((dim1 >> 16) | ((unsigned)Kd << 16));      // dim1 hi | tile_dim0
        g1[4] = (int)16;                               // tile_dim1=16, tile_dim2=0
        g1[5] = (int)(unsigned)(stride0 & 0xFFFFFFFFu);          // dim0_stride lo
        g1[6] = (int)(unsigned)((stride0 >> 32) & 0xFFFFu);      // dim0_stride hi
        g1[7] = 0;
        v4i g2 = {0, 0, 0, 0};                         // group 2 unused (2-D tile)
        v4i g3 = {0, 0, 0, 0};                         // group 3 unused
        v8i g4 = {0, 0, 0, 0, 0, 0, 0, 0};             // clang-23 extra desc group
        __builtin_amdgcn_tensor_load_to_lds(g0, g1, g2, g3, g4, 0);
        __builtin_amdgcn_s_wait_tensorcnt(0);
    }
    __syncthreads();

    v8f acc = {};
    for (int k0 = 0; k0 < Kd; k0 += 32) {
        if (k0 + 32 < Kd) {
            __builtin_prefetch(A + (mt + (lane & 15)) * Kd + k0 + 32, 0, 3);
        }
        v16h a = load_frag_f32(A + (size_t)mt * Kd + k0, Kd, lane);
        v16h b = load_frag_f32(sW + k0, Kd, lane);   // LDS-resident weights
        acc = __builtin_amdgcn_wmma_f32_16x16x32_f16(false, a, false, b,
                                                     (short)0, acc, false, false);
    }
    const int n = nt + (lane & 15);
    const int mbase = mt + ((lane >> 4) << 3);
    const float bv = bias ? bias[n] : 0.0f;
#pragma unroll
    for (int r = 0; r < 8; ++r) {
        float v = acc[r] + bv;
        if (act == 1) v = gelu_exact(v);
        Out[(size_t)(mbase + r) * Nc + n] = v;
    }
}

// -------------------------------------------------------------------------
// Top-C selection of log1p(relu(x)) via full bitonic sort (desc, stable ties)
// -------------------------------------------------------------------------
__global__ void topk_kernel(const float* __restrict__ x,
                            float* __restrict__ vals, float* __restrict__ mz) {
    const int b = blockIdx.x, t = threadIdx.x;  // 1024 threads
    __shared__ float sv[N_];
    __shared__ int   si[N_];
    for (int i = t; i < N_; i += 1024) {
        sv[i] = log1pf(fmaxf(x[b * N_ + i], 0.0f));
        si[i] = i;
    }
    __syncthreads();
    for (int k2 = 2; k2 <= N_; k2 <<= 1) {
        for (int j = k2 >> 1; j > 0; j >>= 1) {
            for (int i = t; i < N_; i += 1024) {
                int ixj = i ^ j;
                if (ixj > i) {
                    float va = sv[i], vb = sv[ixj];
                    int ia = si[i], ib = si[ixj];
                    bool aFirst = (va > vb) || (va == vb && ia < ib);
                    bool desc = ((i & k2) == 0);
                    if (desc ? !aFirst : aFirst) {
                        sv[i] = vb; sv[ixj] = va; si[i] = ib; si[ixj] = ia;
                    }
                }
            }
            __syncthreads();
        }
    }
    for (int i = t; i < C_; i += 1024) {
        vals[b * C_ + i] = sv[i];
        mz[b * C_ + i]   = (float)si[i];
    }
}

// -------------------------------------------------------------------------
// Encoder: roles (PE+emb), reflect-pad conv + point fillers, outer product
// -------------------------------------------------------------------------
__global__ void encoder_kernel(const float* __restrict__ vals,
                               const float* __restrict__ mz,
                               const float* __restrict__ emb_roles,
                               const float* __restrict__ filler_w,
                               const float* __restrict__ conv_w,
                               const float* __restrict__ conv_b,
                               float* __restrict__ outer) {
    const int b = blockIdx.x, c = threadIdx.x;  // 256 threads
    __shared__ float vrow[C_];
    vrow[c] = vals[b * C_ + c];
    __syncthreads();
    float roles[DI_], fill[DI_];
    const float m = mz[b * C_ + c];
    const float neg = -9.210340371976184f / 7.0f;  // -ln(1e4)/(half-1)
#pragma unroll
    for (int d = 0; d < 8; ++d) {
        float ang = m * expf(neg * (float)d);
        roles[d]     = sinf(ang) + emb_roles[c * DI_ + d];
        roles[d + 8] = cosf(ang) + emb_roles[c * DI_ + d + 8];
    }
#pragma unroll
    for (int d = 0; d < DI_; ++d) {
        float acc = conv_b[d];
#pragma unroll
        for (int t = 0; t < 5; ++t) {
            int p = c + t - 2;
            if (p < 0) p = -p;
            if (p >= C_) p = 2 * (C_ - 1) - p;
            acc += conv_w[d * 5 + t] * vrow[p];
        }
        float flocal = gelu_exact(acc);
        fill[d] = gelu_exact(vrow[c] * filler_w[d] + flocal);
    }
    float* orow = outer + (size_t)(b * C_ + c) * (DI_ * DI_);
#pragma unroll
    for (int d = 0; d < DI_; ++d)
#pragma unroll
        for (int e = 0; e < DI_; ++e)
            orow[d * DI_ + e] = roles[d] * fill[e];
}

// -------------------------------------------------------------------------
// LayerNorm over last dim (=128), optional residual add. Block = 128 threads.
// -------------------------------------------------------------------------
__global__ void ln_kernel(const float* __restrict__ X,
                          const float* __restrict__ resid,
                          const float* __restrict__ g,
                          const float* __restrict__ bta,
                          float* __restrict__ Out) {
    const int row = blockIdx.x, t = threadIdx.x;
    __shared__ float rbuf[DM_];
    float v = X[(size_t)row * DM_ + t];
    if (resid) v += resid[(size_t)row * DM_ + t];
    rbuf[t] = v;
    __syncthreads();
    for (int s = 64; s > 0; s >>= 1) { if (t < s) rbuf[t] += rbuf[t + s]; __syncthreads(); }
    const float mean = rbuf[0] / DM_;
    __syncthreads();
    const float d = v - mean;
    rbuf[t] = d * d;
    __syncthreads();
    for (int s = 64; s > 0; s >>= 1) { if (t < s) rbuf[t] += rbuf[t + s]; __syncthreads(); }
    const float var = rbuf[0] / DM_;
    Out[(size_t)row * DM_ + t] = d * rsqrtf(var + 1e-5f) * g[t] + bta[t];
}

// -------------------------------------------------------------------------
// qp/kp projections: qp = q @ W1a^T, kp = k @ W1b^T   (per-head, DH x DH)
// -------------------------------------------------------------------------
__global__ void relproj_kernel(const float* __restrict__ Qg,
                               const float* __restrict__ Kg,
                               const float* __restrict__ w1,
                               float* __restrict__ qp, float* __restrict__ kp) {
    const int idx = blockIdx.x * blockDim.x + threadIdx.x;  // B*H*C*DH
    const int d = idx & (DH_ - 1);
    const int c = (idx >> 5) & (C_ - 1);
    const int bh = idx >> 13;
    const int b = bh >> 2, h = bh & (H_ - 1);
    const float* q  = Qg + (size_t)(b * C_ + c) * DM_ + h * DH_;
    const float* k  = Kg + (size_t)(b * C_ + c) * DM_ + h * DH_;
    const float* wq = w1 + d * (3 * DH_);
    const float* wk = wq + DH_;
    float aq = 0.f, ak = 0.f;
#pragma unroll
    for (int e = 0; e < DH_; ++e) { aq += q[e] * wq[e]; ak += k[e] * wk[e]; }
    qp[idx] = aq;
    kp[idx] = ak;
}

// -------------------------------------------------------------------------
// Second-order scores: one wave per (b,h,i, 16-wide j tile).
// A rows = q_i ⊙ k_j pairs (16x32), B = W3 (32x32) -> 2 WMMAs -> gelu -> w2 dot
// -------------------------------------------------------------------------
__global__ void score_kernel(const float* __restrict__ Qg,
                             const float* __restrict__ Kg,
                             const float* __restrict__ qp,
                             const float* __restrict__ kp,
                             const float* __restrict__ w1,
                             const float* __restrict__ b1v,
                             const float* __restrict__ w2,
                             const float* __restrict__ b2p,
                             float* __restrict__ scores) {
    const int lane = threadIdx.x;
    const int jt = blockIdx.x;       // 0..C/16-1
    const int i  = blockIdx.y;       // 0..C-1
    const int bh = blockIdx.z;       // 0..B*H-1
    const int b = bh / H_, h = bh % H_;
    __shared__ float T[16][32];

    const int r  = lane & 15;
    const int kb = (lane < 16) ? 0 : 8;
    const float* qrow = Qg + (size_t)(b * C_ + i) * DM_ + h * DH_;
    const int jA = jt * 16 + r;
    const float* krow = Kg + (size_t)(b * C_ + jA) * DM_ + h * DH_;

    // 128-bit loads of the q and k sub-rows (16B aligned)
    const float4 q0 = *(const float4*)(qrow + kb);
    const float4 q1 = *(const float4*)(qrow + kb + 4);
    const float4 q2 = *(const float4*)(qrow + kb + 16);
    const float4 q3 = *(const float4*)(qrow + kb + 20);
    const float4 k0 = *(const float4*)(krow + kb);
    const float4 k1 = *(const float4*)(krow + kb + 4);
    const float4 k2 = *(const float4*)(krow + kb + 16);
    const float4 k3 = *(const float4*)(krow + kb + 20);
    v16h a;
    a[0]  = (_Float16)(q0.x * k0.x); a[1]  = (_Float16)(q0.y * k0.y);
    a[2]  = (_Float16)(q0.z * k0.z); a[3]  = (_Float16)(q0.w * k0.w);
    a[4]  = (_Float16)(q1.x * k1.x); a[5]  = (_Float16)(q1.y * k1.y);
    a[6]  = (_Float16)(q1.z * k1.z); a[7]  = (_Float16)(q1.w * k1.w);
    a[8]  = (_Float16)(q2.x * k2.x); a[9]  = (_Float16)(q2.y * k2.y);
    a[10] = (_Float16)(q2.z * k2.z); a[11] = (_Float16)(q2.w * k2.w);
    a[12] = (_Float16)(q3.x * k3.x); a[13] = (_Float16)(q3.y * k3.y);
    a[14] = (_Float16)(q3.z * k3.z); a[15] = (_Float16)(q3.w * k3.w);

#pragma unroll
    for (int nt = 0; nt < 2; ++nt) {
        const float* wrow = w1 + (nt * 16 + r) * (3 * DH_) + 2 * DH_;  // W3[d][:]
        const float4 w0 = *(const float4*)(wrow + kb);
        const float4 wv1 = *(const float4*)(wrow + kb + 4);
        const float4 w2v = *(const float4*)(wrow + kb + 16);
        const float4 w3v = *(const float4*)(wrow + kb + 20);
        v16h bf;
        bf[0]  = (_Float16)w0.x;  bf[1]  = (_Float16)w0.y;
        bf[2]  = (_Float16)w0.z;  bf[3]  = (_Float16)w0.w;
        bf[4]  = (_Float16)wv1.x; bf[5]  = (_Float16)wv1.y;
        bf[6]  = (_Float16)wv1.z; bf[7]  = (_Float16)wv1.w;
        bf[8]  = (_Float16)w2v.x; bf[9]  = (_Float16)w2v.y;
        bf[10] = (_Float16)w2v.z; bf[11] = (_Float16)w2v.w;
        bf[12] = (_Float16)w3v.x; bf[13] = (_Float16)w3v.y;
        bf[14] = (_Float16)w3v.z; bf[15] = (_Float16)w3v.w;
        v8f c = {};
        c = __builtin_amdgcn_wmma_f32_16x16x32_f16(false, a, false, bf,
                                                   (short)0, c, false, false);
        const int mb = (lane >> 4) << 3;
#pragma unroll
        for (int rr = 0; rr < 8; ++rr) T[mb + rr][nt * 16 + r] = c[rr];
    }
    __syncthreads();

    // epilogue: two lanes per pair, each half of DH, shuffle-reduce
    const int p   = lane & 15;
    const int dh0 = (lane < 16) ? 0 : 16;
    const int j   = jt * 16 + p;
    const float* qpr = qp + (size_t)(bh * C_ + i) * DH_;
    const float* kpr = kp + (size_t)(bh * C_ + j) * DH_;
    float partial = 0.f;
#pragma unroll
    for (int d = 0; d < 16; ++d) {
        const int dd = dh0 + d;
        float res = T[p][dd] + qpr[dd] + kpr[dd] + b1v[dd];
        partial += w2[dd] * gelu_exact(res);
    }
    partial += __shfl_xor(partial, 16, 32);
    if (lane < 16) {
        scores[((size_t)bh * C_ + i) * C_ + j] =
            (partial + b2p[0]) * 0.17677669529663688f;  // 1/sqrt(DH)
    }
}

// -------------------------------------------------------------------------
// Per-row sparse-top-K softmax + attn@V. Block per (b,h,i), 256 threads.
// -------------------------------------------------------------------------
__global__ void attn_kernel(const float* __restrict__ scores,
                            const float* __restrict__ Vg,
                            float* __restrict__ attnout) {
    const int i = blockIdx.x, bh = blockIdx.y;
    const int b = bh / H_, h = bh % H_;
    const int t = threadIdx.x;
    __shared__ float srt[C_];
    __shared__ float attn[C_];
    const float* srow = scores + ((size_t)bh * C_ + i) * C_;
    const float myv = srow[t];
    srt[t] = myv;
    __syncthreads();
    // bitonic sort, descending
    for (int k2 = 2; k2 <= C_; k2 <<= 1) {
        for (int j = k2 >> 1; j > 0; j >>= 1) {
            int ixj = t ^ j;
            if (ixj > t) {
                float va = srt[t], vb = srt[ixj];
                bool desc = ((t & k2) == 0);
                if (desc ? (va < vb) : (va > vb)) { srt[t] = vb; srt[ixj] = va; }
            }
            __syncthreads();
        }
    }
    const float minv = srt[K_ - 1];
    const float vmax = srt[0];
    __syncthreads();
    const float e = (myv >= minv) ? expf(myv - vmax) : 0.0f;
    attn[t] = e;
    srt[t] = e;
    __syncthreads();
    for (int s = 128; s > 0; s >>= 1) { if (t < s) srt[t] += srt[t + s]; __syncthreads(); }
    const float inv = 1.0f / srt[0];
    if (t < DH_) {
        float acc = 0.f;
        for (int j = 0; j < C_; ++j)
            acc += attn[j] * Vg[(size_t)(b * C_ + j) * DM_ + h * DH_ + t];
        attnout[(size_t)(b * C_ + i) * DM_ + h * DH_ + t] = acc * inv;
    }
}

// -------------------------------------------------------------------------
// Task-query readout pooling. Block per batch, 256 threads.
// -------------------------------------------------------------------------
__global__ void pool_kernel(const float* __restrict__ bnd,
                            const float* __restrict__ tq,
                            float* __restrict__ hout) {
    const int b = blockIdx.x, t = threadIdx.x;
    __shared__ float sc[C_];
    __shared__ float red[C_];
    const float* row = bnd + (size_t)(b * C_ + t) * DM_;
    float acc = 0.f;
    for (int d = 0; d < DM_; ++d) acc += tq[d] * row[d];
    sc[t] = acc * 0.08838834764831845f;  // 1/sqrt(DM)
    red[t] = sc[t];
    __syncthreads();
    for (int s = 128; s > 0; s >>= 1) { if (t < s) red[t] = fmaxf(red[t], red[t + s]); __syncthreads(); }
    const float mx = red[0];
    __syncthreads();
    const float e = expf(sc[t] - mx);
    sc[t] = e; red[t] = e;
    __syncthreads();
    for (int s = 128; s > 0; s >>= 1) { if (t < s) red[t] += red[t + s]; __syncthreads(); }
    const float inv = 1.0f / red[0];
    __syncthreads();
    if (t < DM_) {
        float a = 0.f;
        for (int c = 0; c < C_; ++c) a += sc[c] * bnd[(size_t)(b * C_ + c) * DM_ + t];
        hout[b * DM_ + t] = a * inv;
    }
}

// -------------------------------------------------------------------------
// Gated relational memory: slot attention + GRU. Block per (b,s), 256 threads.
// -------------------------------------------------------------------------
__global__ void memory_kernel(const float* __restrict__ bnd,
                              const float* __restrict__ slots,
                              const float* __restrict__ wih,
                              const float* __restrict__ whh,
                              const float* __restrict__ bih,
                              const float* __restrict__ bhh,
                              float* __restrict__ mstate) {
    const int s = blockIdx.x, b = blockIdx.y, t = threadIdx.x;
    __shared__ float prob[C_];
    __shared__ float red[C_];
    __shared__ float ctx[DM_];
    __shared__ float gi[3 * DM_];
    __shared__ float gh[3 * DM_];
    const float* slot = slots + s * DM_;
    {
        const float* row = bnd + (size_t)(b * C_ + t) * DM_;
        float acc = 0.f;
        for (int d = 0; d < DM_; ++d) acc += slot[d] * row[d];
        prob[t] = acc * 0.08838834764831845f;
        red[t] = prob[t];
    }
    __syncthreads();
    for (int r = 128; r > 0; r >>= 1) { if (t < r) red[t] = fmaxf(red[t], red[t + r]); __syncthreads(); }
    const float mx = red[0];
    __syncthreads();
    const float e = expf(prob[t] - mx);
    prob[t] = e; red[t] = e;
    __syncthreads();
    for (int r = 128; r > 0; r >>= 1) { if (t < r) red[t] += red[t + r]; __syncthreads(); }
    const float inv = 1.0f / red[0];
    __syncthreads();
    if (t < DM_) {
        float a = 0.f;
        for (int c = 0; c < C_; ++c) a += prob[c] * bnd[(size_t)(b * C_ + c) * DM_ + t];
        ctx[t] = a * inv;
    }
    __syncthreads();
    for (int g = t; g < 3 * DM_; g += 256) {
        float ai = bih[g], ah = bhh[g];
        for (int d = 0; d < DM_; ++d) {
            ai += wih[g * DM_ + d] * ctx[d];
            ah += whh[g * DM_ + d] * slot[d];
        }
        gi[g] = ai; gh[g] = ah;
    }
    __syncthreads();
    if (t < DM_) {
        float r = 1.f / (1.f + expf(-(gi[t] + gh[t])));
        float z = 1.f / (1.f + expf(-(gi[DM_ + t] + gh[DM_ + t])));
        float n = tanhf(gi[2 * DM_ + t] + r * gh[2 * DM_ + t]);
        mstate[(size_t)(b * S_ + s) * DM_ + t] = (1.f - z) * n + z * slot[t];
    }
}

// -------------------------------------------------------------------------
// mem = LN(mean_s(mstate)); out = (h+mem) @ head_w^T + head_b
// -------------------------------------------------------------------------
__global__ void head_kernel(const float* __restrict__ mstate,
                            const float* __restrict__ hbuf,
                            const float* __restrict__ mg,
                            const float* __restrict__ mb,
                            const float* __restrict__ hw,
                            const float* __restrict__ hb,
                            float* __restrict__ out) {
    const int b = blockIdx.x, t = threadIdx.x;  // 128 threads
    __shared__ float mem[DM_];
    __shared__ float rbuf[DM_];
    float acc = 0.f;
    for (int s = 0; s < S_; ++s) acc += mstate[(size_t)(b * S_ + s) * DM_ + t];
    acc *= (1.0f / S_);
    rbuf[t] = acc;
    __syncthreads();
    for (int r = 64; r > 0; r >>= 1) { if (t < r) rbuf[t] += rbuf[t + r]; __syncthreads(); }
    const float mean = rbuf[0] / DM_;
    __syncthreads();
    const float d = acc - mean;
    rbuf[t] = d * d;
    __syncthreads();
    for (int r = 64; r > 0; r >>= 1) { if (t < r) rbuf[t] += rbuf[t + r]; __syncthreads(); }
    const float var = rbuf[0] / DM_;
    mem[t] = d * rsqrtf(var + 1e-5f) * mg[t] + mb[t] + hbuf[b * DM_ + t];
    __syncthreads();
    if (t < OUT_) {
        float o = hb[t];
        for (int d2 = 0; d2 < DM_; ++d2) o += mem[d2] * hw[t * DM_ + d2];
        out[b * OUT_ + t] = o;
    }
}

// -------------------------------------------------------------------------
extern "C" void kernel_launch(void* const* d_in, const int* in_sizes, int n_in,
                              void* d_out, int out_size, void* d_ws, size_t ws_size,
                              hipStream_t stream) {
    (void)in_sizes; (void)n_in; (void)out_size; (void)ws_size;
    const float* x         = (const float*)d_in[0];
    const float* emb_roles = (const float*)d_in[1];
    const float* filler_w  = (const float*)d_in[2];
    const float* conv_w    = (const float*)d_in[3];
    const float* conv_b    = (const float*)d_in[4];
    const float* outer_w   = (const float*)d_in[5];
    const float* enc_ng    = (const float*)d_in[6];
    const float* enc_nb    = (const float*)d_in[7];
    const float* qw        = (const float*)d_in[8];
    const float* qb        = (const float*)d_in[9];
    const float* kw        = (const float*)d_in[10];
    const float* kb        = (const float*)d_in[11];
    const float* vw        = (const float*)d_in[12];
    const float* vb        = (const float*)d_in[13];
    const float* rel_w1    = (const float*)d_in[14];
    const float* rel_b1    = (const float*)d_in[15];
    const float* rel_w2    = (const float*)d_in[16];
    const float* rel_b2    = (const float*)d_in[17];
    const float* ow        = (const float*)d_in[18];
    const float* ob        = (const float*)d_in[19];
    const float* n1g       = (const float*)d_in[20];
    const float* n1b       = (const float*)d_in[21];
    const float* n2g       = (const float*)d_in[22];
    const float* n2b       = (const float*)d_in[23];
    const float* ffn_w1    = (const float*)d_in[24];
    const float* ffn_b1    = (const float*)d_in[25];
    const float* ffn_w2    = (const float*)d_in[26];
    const float* ffn_b2    = (const float*)d_in[27];
    const float* task_q    = (const float*)d_in[28];
    const float* slots     = (const float*)d_in[29];
    const float* gru_wih   = (const float*)d_in[30];
    const float* gru_whh   = (const float*)d_in[31];
    const float* gru_bih   = (const float*)d_in[32];
    const float* gru_bhh   = (const float*)d_in[33];
    const float* mem_ng    = (const float*)d_in[34];
    const float* mem_nb    = (const float*)d_in[35];
    const float* head_w    = (const float*)d_in[36];
    const float* head_b    = (const float*)d_in[37];

    float* w = (float*)d_ws;
    size_t off = 0;
    auto alloc = [&](size_t n) { float* p = w + off; off += n; return p; };
    float* vals    = alloc(B_ * C_);
    float* mz      = alloc(B_ * C_);
    float* outer   = alloc((size_t)B_ * C_ * DI_ * DI_);
    float* bnd     = alloc((size_t)B_ * C_ * DM_);
    float* Qb      = alloc((size_t)B_ * C_ * DM_);
    float* Kb      = alloc((size_t)B_ * C_ * DM_);
    float* Vb      = alloc((size_t)B_ * C_ * DM_);
    float* qp      = alloc((size_t)B_ * H_ * C_ * DH_);
    float* kp      = alloc((size_t)B_ * H_ * C_ * DH_);
    float* scores  = alloc((size_t)B_ * H_ * C_ * C_);
    float* attnout = alloc((size_t)B_ * C_ * DM_);
    float* tmp1    = alloc((size_t)B_ * C_ * 4 * DM_);
    float* tmp2    = alloc((size_t)B_ * C_ * DM_);
    float* b1buf   = alloc((size_t)B_ * C_ * DM_);
    float* hbuf    = alloc(B_ * DM_);
    float* mstate  = alloc((size_t)B_ * S_ * DM_);

    const int M = B_ * C_;  // 512 rows

    // top-k + encoder
    topk_kernel<<<B_, 1024, 0, stream>>>(x, vals, mz);
    encoder_kernel<<<B_, C_, 0, stream>>>(vals, mz, emb_roles, filler_w,
                                          conv_w, conv_b, outer);
    wmma_linear_kernel<<<dim3(DM_ / 16, M / 16), 32, 0, stream>>>(
        outer, outer_w, nullptr, tmp2, M, DI_ * DI_, DM_, 0);
    ln_kernel<<<M, DM_, 0, stream>>>(tmp2, nullptr, enc_ng, enc_nb, bnd);

    for (int i = 0; i < L_; ++i) {
        const float* qw_l = qw + (size_t)i * DM_ * DM_;
        const float* kw_l = kw + (size_t)i * DM_ * DM_;
        const float* vw_l = vw + (size_t)i * DM_ * DM_;
        const float* ow_l = ow + (size_t)i * DM_ * DM_;
        const float* w1_l = rel_w1 + (size_t)i * DH_ * 3 * DH_;
        const float* f1_l = ffn_w1 + (size_t)i * 4 * DM_ * DM_;
        const float* f2_l = ffn_w2 + (size_t)i * DM_ * 4 * DM_;

        wmma_linear_kernel<<<dim3(DM_ / 16, M / 16), 32, 0, stream>>>(
            bnd, qw_l, qb + i * DM_, Qb, M, DM_, DM_, 0);
        wmma_linear_kernel<<<dim3(DM_ / 16, M / 16), 32, 0, stream>>>(
            bnd, kw_l, kb + i * DM_, Kb, M, DM_, DM_, 0);
        wmma_linear_kernel<<<dim3(DM_ / 16, M / 16), 32, 0, stream>>>(
            bnd, vw_l, vb + i * DM_, Vb, M, DM_, DM_, 0);

        relproj_kernel<<<(B_ * H_ * C_ * DH_) / 256, 256, 0, stream>>>(
            Qb, Kb, w1_l, qp, kp);

        score_kernel<<<dim3(C_ / 16, C_, B_ * H_), 32, 0, stream>>>(
            Qb, Kb, qp, kp, w1_l, rel_b1 + i * DH_, rel_w2 + i * DH_,
            rel_b2 + i, scores);

        attn_kernel<<<dim3(C_, B_ * H_), C_, 0, stream>>>(scores, Vb, attnout);

        wmma_linear_kernel<<<dim3(DM_ / 16, M / 16), 32, 0, stream>>>(
            attnout, ow_l, ob + i * DM_, tmp2, M, DM_, DM_, 0);
        ln_kernel<<<M, DM_, 0, stream>>>(tmp2, bnd, n1g + i * DM_, n1b + i * DM_, b1buf);

        wmma_linear_kernel<<<dim3(4 * DM_ / 16, M / 16), 32, 0, stream>>>(
            b1buf, f1_l, ffn_b1 + i * 4 * DM_, tmp1, M, DM_, 4 * DM_, 1);
        wmma_linear_kernel<<<dim3(DM_ / 16, M / 16), 32, 0, stream>>>(
            tmp1, f2_l, ffn_b2 + i * DM_, tmp2, M, 4 * DM_, DM_, 0);
        ln_kernel<<<M, DM_, 0, stream>>>(tmp2, b1buf, n2g + i * DM_, n2b + i * DM_, bnd);
    }

    pool_kernel<<<B_, C_, 0, stream>>>(bnd, task_q, hbuf);
    memory_kernel<<<dim3(S_, B_), C_, 0, stream>>>(bnd, slots, gru_wih, gru_whh,
                                                   gru_bih, gru_bhh, mstate);
    head_kernel<<<B_, DM_, 0, stream>>>(mstate, hbuf, mem_ng, mem_nb,
                                        head_w, head_b, (float*)d_out);
}
